// ConvAttnPool_73804718014831
// MI455X (gfx1250) — matrix operand order repaired and checked by reading, not compiled, not run on hardware
//
#include <hip/hip_runtime.h>
#include <hip/hip_bf16.h>

typedef __bf16 bf16_t;
typedef __attribute__((ext_vector_type(16))) __bf16 v16bf;
typedef __attribute__((ext_vector_type(8)))  float  v8f;
typedef unsigned int u32x4 __attribute__((ext_vector_type(4)));
typedef int          i32x4 __attribute__((ext_vector_type(4)));
typedef int          i32x8 __attribute__((ext_vector_type(8)));

union BF16x16 { v16bf v; bf16_t e[16]; };
union F32x8   { v8f   v; float  e[8];  };

#define Bb   8
#define Ls   2500
#define Es   100
#define Fs   50
#define Ks   9
#define Ys   8922
#define NYT  558      // ceil(Y/16)
#define NLT  79       // ceil(L/32)
#define KPAD 8928     // ceil(Y/32)*32
#define NCOL 400      // B*F
#define NSTEP (KPAD / 32)   // 279

#ifndef __has_builtin
#define __has_builtin(x) 0
#endif
#if __has_builtin(__builtin_amdgcn_tensor_load_to_lds) && \
    __has_builtin(__builtin_amdgcn_s_wait_tensorcnt)
#define HAVE_TDM 1
#else
#define HAVE_TDM 0
#endif

// ---------------------------------------------------------------------------
// TDM helper: 2D tile load Global->LDS, 16-bit elements.
// D# bit layout per CDNA5 ISA ch.8 (group0: count/lds_addr/global_addr/type=2,
// group1: data_size=2B, tensor dims, tile dims, dim0 stride).
// ---------------------------------------------------------------------------
#if HAVE_TDM
__device__ __forceinline__ void tdm_load_2d_bf16(
    unsigned lds_addr, const void* gaddr,
    unsigned tile_d0, unsigned tile_d1,
    unsigned tensor_d0, unsigned tensor_d1,
    unsigned long long d0_stride)
{
    unsigned long long ga = (unsigned long long)(uintptr_t)gaddr;
    u32x4 g0;
    g0[0] = 1u;                                          // count=1, load, no gather
    g0[1] = lds_addr;                                    // LDS byte address
    g0[2] = (unsigned)(ga & 0xffffffffu);                // global_addr[31:0]
    g0[3] = (unsigned)((ga >> 32) & 0x01ffffffu) | (2u << 30);   // addr[56:32]|type=2
    i32x8 g1;
    g1[0] = (int)(1u << 16);                             // data_size=1 (2 bytes)
    g1[1] = (int)(tensor_d0 << 16);                      // tensor_dim0[15:0] @bit48
    g1[2] = (int)((tensor_d0 >> 16) | (tensor_d1 << 16));// dim0 hi | dim1 lo
    g1[3] = (int)((tensor_d1 >> 16) | (tile_d0 << 16));  // dim1 hi | tile_dim0
    g1[4] = (int)(tile_d1 & 0xffffu);                    // tile_dim1 (tile_dim2=0)
    g1[5] = (int)(unsigned)(d0_stride & 0xffffffffu);    // tensor_dim0_stride lo
    g1[6] = (int)(unsigned)((d0_stride >> 32) & 0xffffu);// stride hi (dim1_stride=0)
    g1[7] = 0;
    i32x4 z4 = {0, 0, 0, 0};
#if __clang_major__ >= 23
    i32x8 z8 = {0, 0, 0, 0, 0, 0, 0, 0};
    __builtin_amdgcn_tensor_load_to_lds(g0, g1, z4, z4, z8, 0);
#else
    __builtin_amdgcn_tensor_load_to_lds(g0, g1, z4, z4, 0);
#endif
}
#endif

// ---------------------------------------------------------------------------
// Kernel 1: embedding gather + conv1d(K=9, same padding) + tanh -> bf16
// hp layout: [B][L][64]  (cols 50..63 zero-padded for WMMA K-padding)
// ---------------------------------------------------------------------------
__global__ __launch_bounds__(256)
void embconv_kernel(const int* __restrict__ x, const float* __restrict__ Ew,
                    const float* __restrict__ Cw, const float* __restrict__ Cb,
                    bf16_t* __restrict__ hp)
{
    __shared__ float es[72 * Es];                  // 72 rows x 100 cols f32
    const int b    = blockIdx.y;
    const int base = blockIdx.x * 64;
    const int tid  = threadIdx.x;

    for (int idx = tid; idx < 72 * Es; idx += 256) {
        int row = idx / Es, col = idx % Es;
        int l = base - 4 + row;
        float v = 0.0f;
        if (l >= 0 && l < Ls) {
            int xi = x[b * Ls + l];
            v = Ew[(size_t)xi * Es + col];
        }
        es[idx] = v;
    }
    __syncthreads();

    const int lloc = tid & 63;
    const int f0   = tid >> 6;                     // 0..3
    const int l    = base + lloc;
    if (l < Ls) {
        bf16_t* dst = hp + ((size_t)b * Ls + l) * 64;
        for (int f = f0; f < Fs; f += 4) {
            float a = Cb[f];
            const float* wf = Cw + f * (Es * Ks);
            for (int e = 0; e < Es; ++e) {
                const float* we = wf + e * Ks;
                #pragma unroll
                for (int k = 0; k < Ks; ++k)
                    a += es[(lloc + k) * Es + e] * we[k];
            }
            dst[f] = (bf16_t)tanhf(a);
        }
        for (int f = Fs + f0; f < 64; f += 4) dst[f] = (bf16_t)0.0f;
    }
}

// ---------------------------------------------------------------------------
// Kernel 2: U4_w f32 [Y][50] -> bf16 [Y][64] zero-padded
// ---------------------------------------------------------------------------
__global__ __launch_bounds__(256)
void u4cvt_kernel(const float* __restrict__ U4, bf16_t* __restrict__ U4b)
{
    int idx = blockIdx.x * 256 + threadIdx.x;
    if (idx >= Ys * 64) return;
    int y = idx >> 6, f = idx & 63;
    U4b[idx] = (f < Fs) ? (bf16_t)U4[y * Fs + f] : (bf16_t)0.0f;
}

// ---------------------------------------------------------------------------
// Kernel 3: flash-style per-label attention.
//   scores[b,y,l] = sum_f U4[y,f] hp[b,l,f];  alpha = softmax_l;
//   m4t[b,y,f]    = sum_l alpha * hp[b,l,f]
// 4 waves/block, each wave owns one 16-row label tile; hp L-tile (32x64 bf16)
// shared in LDS.  8 v_wmma_f32_16x16x32_bf16 per wave per L-step.
// ---------------------------------------------------------------------------
__global__ __launch_bounds__(128)
void attn_kernel(const bf16_t* __restrict__ hp,     // [B][L][64]
                 const bf16_t* __restrict__ u4b,    // [Y][64]
                 float* __restrict__ m4t)           // [B][Y][50]
{
    __shared__ bf16_t hp_tile[32 * 72];   // [l][f]    (score B operand)
    __shared__ bf16_t hp_tT [64 * 40];    // [f][l]    (m4t   B operand)
    __shared__ bf16_t Pw    [4 * 16 * 40];// per-wave prob tile [y][l]

    const int tid   = threadIdx.x;
    const int ln    = tid & 31;
    const int w     = tid >> 5;
    const int b     = blockIdx.y;
    const int yt    = blockIdx.x * 4 + w;
    const bool act  = (yt < NYT);
    const int ybase = yt * 16;
    const int g     = ln >> 4;            // lane half-group
    const int lx    = ln & 15;

    // U4 A-fragments for K = [0,32) and [32,64)
    BF16x16 a4[2];
    {
        int y = ybase + lx;
        #pragma unroll
        for (int kf = 0; kf < 2; ++kf) {
            if (act && y < Ys) {
                const bf16_t* src = u4b + (size_t)y * 64 + kf * 32 + g * 8;
                *reinterpret_cast<int4*>(&a4[kf].e[0]) =
                    *reinterpret_cast<const int4*>(src);
                *reinterpret_cast<int4*>(&a4[kf].e[8]) =
                    *reinterpret_cast<const int4*>(src + 16);
            } else {
                #pragma unroll
                for (int e = 0; e < 16; ++e) a4[kf].e[e] = (bf16_t)0.0f;
            }
        }
    }

    float Mrun[8], Srun[8];
    #pragma unroll
    for (int r = 0; r < 8; ++r) { Mrun[r] = -1e30f; Srun[r] = 0.0f; }
    F32x8 acc[4];
    #pragma unroll
    for (int t = 0; t < 4; ++t) { v8f z = {}; acc[t].v = z; }

    bf16_t* Pme = &Pw[w * 16 * 40];

    for (int lt = 0; lt < NLT; ++lt) {
        __syncthreads();                               // WAR on hp tiles
        {   // cooperative stage of hp[b, lt*32 .. +31, 0..63] (+ transpose)
            const size_t gbase = ((size_t)b * Ls + (size_t)lt * 32) * 64;
            // prefetch next L-tile (global_prefetch_b8): 128 thr x 32B = 4KB
            if (lt + 1 < NLT)
                __builtin_prefetch((const char*)(hp + gbase + 2048) + tid * 32, 0, 0);
            for (int idx = tid; idx < 256; idx += 128) {
                int row = idx >> 3, c16 = idx & 7;
                int l = lt * 32 + row;
                int4 vq;
                if (l < Ls)
                    vq = *reinterpret_cast<const int4*>(hp + gbase + row * 64 + c16 * 8);
                else
                    vq = make_int4(0, 0, 0, 0);
                *reinterpret_cast<int4*>(&hp_tile[row * 72 + c16 * 8]) = vq;
                const bf16_t* hv = reinterpret_cast<const bf16_t*>(&vq);
                #pragma unroll
                for (int j = 0; j < 8; ++j)
                    hp_tT[(c16 * 8 + j) * 40 + row] = hv[j];
            }
        }
        __syncthreads();                               // RAW

        if (act) {
            // ---- score GEMM: two 16x16 D tiles (l halves), K=64 (2 steps)
            F32x8 sc[2];
            #pragma unroll
            for (int h = 0; h < 2; ++h) {
                v8f z = {}; sc[h].v = z;
                #pragma unroll
                for (int kf = 0; kf < 2; ++kf) {
                    BF16x16 bF;
                    const bf16_t* src = &hp_tile[(h * 16 + lx) * 72 + kf * 32 + g * 8];
                    *reinterpret_cast<int4*>(&bF.e[0]) = *reinterpret_cast<const int4*>(src);
                    *reinterpret_cast<int4*>(&bF.e[8]) = *reinterpret_cast<const int4*>(src + 16);
                    sc[h].v = __builtin_amdgcn_wmma_f32_16x16x32_bf16(
                        false, a4[kf].v, false, bF.v, (short)0, sc[h].v, false, false);
                }
            }
            // mask l >= L
            #pragma unroll
            for (int h = 0; h < 2; ++h) {
                int l = lt * 32 + h * 16 + lx;
                if (l >= Ls) {
                    #pragma unroll
                    for (int r = 0; r < 8; ++r) sc[h].e[r] = -1e30f;
                }
            }
            // ---- online softmax (row = r + 8*g, cols spread over 16 lanes)
            #pragma unroll
            for (int r = 0; r < 8; ++r) {
                float mx = fmaxf(sc[0].e[r], sc[1].e[r]);
                #pragma unroll
                for (int msk = 8; msk >= 1; msk >>= 1)
                    mx = fmaxf(mx, __shfl_xor(mx, msk, 32));
                float newM = fmaxf(Mrun[r], mx);
                float corr = __expf(Mrun[r] - newM);
                float p0 = __expf(sc[0].e[r] - newM);
                float p1 = __expf(sc[1].e[r] - newM);
                float ps = p0 + p1;
                #pragma unroll
                for (int msk = 8; msk >= 1; msk >>= 1)
                    ps += __shfl_xor(ps, msk, 32);
                Srun[r] = Srun[r] * corr + ps;
                Mrun[r] = newM;
                #pragma unroll
                for (int t = 0; t < 4; ++t) acc[t].e[r] *= corr;
                int m = r + 8 * g;
                Pme[m * 40 + lx]      = (bf16_t)p0;     // intra-wave LDS: in-order
                Pme[m * 40 + 16 + lx] = (bf16_t)p1;
            }
            // ---- m4t GEMM: acc[16x64] += P[16x32] @ hp_tile[32x64]
            BF16x16 pA;
            {
                const bf16_t* src = &Pme[lx * 40 + g * 8];
                *reinterpret_cast<int4*>(&pA.e[0]) = *reinterpret_cast<const int4*>(src);
                *reinterpret_cast<int4*>(&pA.e[8]) = *reinterpret_cast<const int4*>(src + 16);
            }
            #pragma unroll
            for (int t = 0; t < 4; ++t) {
                BF16x16 bF;
                const bf16_t* src = &hp_tT[(t * 16 + lx) * 40 + g * 8];
                *reinterpret_cast<int4*>(&bF.e[0]) = *reinterpret_cast<const int4*>(src);
                *reinterpret_cast<int4*>(&bF.e[8]) = *reinterpret_cast<const int4*>(src + 16);
                acc[t].v = __builtin_amdgcn_wmma_f32_16x16x32_bf16(
                    false, pA.v, false, bF.v, (short)0, acc[t].v, false, false);
            }
        }
    }

    if (act) {
        #pragma unroll
        for (int t = 0; t < 4; ++t) {
            int f = t * 16 + lx;
            if (f < Fs) {
                #pragma unroll
                for (int r = 0; r < 8; ++r) {
                    int y = ybase + r + 8 * g;
                    if (y < Ys)
                        m4t[((size_t)b * Ys + y) * Fs + f] = acc[t].e[r] / Srun[r];
                }
            }
        }
    }
}

// ---------------------------------------------------------------------------
// Kernel 4: support[b,z,g] = sum_f m4t[b,z,f] * gcn_w[f,g], stored TRANSPOSED
// bf16: SbT[n][z], n = b*50+g, cols z in [Y, 8928) zero-filled (K padding).
// Transposed layout makes adj-GEMM B tiles natural [n][k] 2D blocks (TDM-able).
// ---------------------------------------------------------------------------
__global__ __launch_bounds__(256)
void support_kernel(const float* __restrict__ m4t, const float* __restrict__ Gw,
                    bf16_t* __restrict__ SbT)
{
    int idx = blockIdx.x * 256 + threadIdx.x;      // exactly 400*8928 threads
    int n = idx / KPAD, z = idx % KPAD;            // adjacent tid -> adjacent z
    float a = 0.0f;
    if (z < Ys) {
        int b = n / Fs, gcol = n % Fs;
        const float* mr = m4t + ((size_t)b * Ys + z) * Fs;
        #pragma unroll 5
        for (int f = 0; f < Fs; ++f) a += mr[f] * Gw[f * Fs + gcol];
    }
    SbT[idx] = (bf16_t)a;
}

// ---------------------------------------------------------------------------
// Kernel 5: out1raw[y][n] = sum_z adj[y,z] * support[z][n]   (Y x 8928 x 400)
// adj converted f32->bf16 while staging A tiles.  B tile = [n=400][k=32]
// sub-block of SbT: double-buffered in LDS, fetched by the Tensor Data Mover
// (TENSOR_LOAD_TO_LDS + s_wait_tensorcnt) with the next tile's DMA in flight
// during the current tile's 25 WMMAs; manual-staging fallback if no TDM.
// ---------------------------------------------------------------------------
__global__ __launch_bounds__(128)
void adjgemm_kernel(const float* __restrict__ adj,   // [Y][Y] f32
                    const bf16_t* __restrict__ SbT,  // [400][8928] bf16
                    float* __restrict__ out1)        // [Y][400] f32 (pre-act)
{
    __shared__ bf16_t Atile[16 * 40];                // [m][k]
    __shared__ bf16_t Btile[2][NCOL * 32];           // [n][k], double buffered
    const int tid = threadIdx.x, ln = tid & 31, w = tid >> 5;
    const int g = ln >> 4, lx = ln & 15;
    const int mbase = blockIdx.x * 16;

    F32x8 acc[7];
    #pragma unroll
    for (int i = 0; i < 7; ++i) { v8f z = {}; acc[i].v = z; }

#if HAVE_TDM
    if (w == 0) {
        tdm_load_2d_bf16((unsigned)(uintptr_t)&Btile[0][0], SbT,
                         /*tile*/32, NCOL, /*tensor*/KPAD, NCOL,
                         (unsigned long long)KPAD);
    }
#endif

    for (int s = 0; s < NSTEP; ++s) {
        const int kb  = s * 32;
        const int cur = s & 1;
        __syncthreads();   // all waves done reading previous A tile + B buffer

        // stage A: adj[mbase..+15][kb..+31] f32 -> bf16 (+ prefetch next)
        if (kb + 32 < KPAD) {
            int py = mbase + (tid >> 3);
            if (py < Ys)
                __builtin_prefetch(adj + (size_t)py * Ys + (kb + 32) + (tid & 7) * 4,
                                   0, 0);
        }
        for (int idx = tid; idx < 512; idx += 128) {
            int row = idx >> 5, col = idx & 31;
            int y = mbase + row, k = kb + col;
            float v = (y < Ys && k < Ys) ? adj[(size_t)y * Ys + k] : 0.0f;
            Atile[row * 40 + col] = (bf16_t)v;
        }

#if HAVE_TDM
        if (w == 0) {
            __builtin_amdgcn_s_wait_tensorcnt((short)0);   // B tile s landed
            if (s + 1 < NSTEP)
                tdm_load_2d_bf16((unsigned)(uintptr_t)&Btile[(s + 1) & 1][0],
                                 SbT + (kb + 32),
                                 /*tile*/32, NCOL,
                                 /*tensor*/(unsigned)(KPAD - (kb + 32)), NCOL,
                                 (unsigned long long)KPAD);
        }
#else
        // manual staging fallback: coalesced int4 copy, no transpose needed
        for (int idx = tid; idx < 1600; idx += 128) {
            int n = idx >> 2, ch = idx & 3;
            int4 vq = *reinterpret_cast<const int4*>(
                SbT + (size_t)n * KPAD + kb + ch * 8);
            *reinterpret_cast<int4*>(&Btile[cur][n * 32 + ch * 8]) = vq;
        }
#endif
        __syncthreads();   // A staged, B tile s visible to all waves

        BF16x16 aF;
        {
            const bf16_t* src = &Atile[lx * 40 + g * 8];
            *reinterpret_cast<int4*>(&aF.e[0]) = *reinterpret_cast<const int4*>(src);
            *reinterpret_cast<int4*>(&aF.e[8]) = *reinterpret_cast<const int4*>(src + 16);
        }
        #pragma unroll
        for (int i = 0; i < 7; ++i) {
            int nt = w + 4 * i;
            if (nt < 25) {
                BF16x16 bF;
                const bf16_t* src = &Btile[cur][(nt * 16 + lx) * 32 + g * 8];
                *reinterpret_cast<int4*>(&bF.e[0]) = *reinterpret_cast<const int4*>(src);
                *reinterpret_cast<int4*>(&bF.e[8]) = *reinterpret_cast<const int4*>(src + 16);
                acc[i].v = __builtin_amdgcn_wmma_f32_16x16x32_bf16(
                    false, aF.v, false, bF.v, (short)0, acc[i].v, false, false);
            }
        }
    }

    #pragma unroll
    for (int i = 0; i < 7; ++i) {
        int nt = w + 4 * i;
        if (nt < 25) {
            int n = nt * 16 + lx;
            #pragma unroll
            for (int r = 0; r < 8; ++r) {
                int y = mbase + r + 8 * g;
                if (y < Ys) out1[(size_t)y * NCOL + n] = acc[i].e[r];
            }
        }
    }
}

// ---------------------------------------------------------------------------
// Kernel 6: epilogue — bias + leaky_relu on out1, label-wise dots -> y4t, y4
// d_out = [ y4t (B*Y) | y4 (B*Y) ]
// ---------------------------------------------------------------------------
__global__ __launch_bounds__(256)
void final_kernel(const float* __restrict__ m4t, const float* __restrict__ out1,
                  const float* __restrict__ gcn_b,
                  const float* __restrict__ f4t_w, const float* __restrict__ f4t_b,
                  const float* __restrict__ f4_w,  const float* __restrict__ f4_b,
                  float* __restrict__ out)
{
    int idx = blockIdx.x * 256 + threadIdx.x;
    if (idx >= Bb * Ys) return;
    int b = idx / Ys, y = idx % Ys;
    const float* mr = m4t + ((size_t)b * Ys + y) * Fs;
    const float* o1 = out1 + (size_t)y * NCOL + b * Fs;
    float y4t = f4t_b[y], y4 = f4_b[y];
    #pragma unroll 5
    for (int f = 0; f < Fs; ++f) {
        float m = mr[f];
        y4t += m * f4t_w[(size_t)y * Fs + f];
        y4  += m * f4_w[(size_t)y * (2 * Fs) + f];
    }
    #pragma unroll 5
    for (int gg = 0; gg < Fs; ++gg) {
        float o = o1[gg] + gcn_b[gg];
        o = (o > 0.0f) ? o : 0.2f * o;          // leaky_relu 0.2
        y4 += o * f4_w[(size_t)y * (2 * Fs) + Fs + gg];
    }
    out[(size_t)b * Ys + y] = y4t;
    out[(size_t)Bb * Ys + (size_t)b * Ys + y] = y4;
}

// ---------------------------------------------------------------------------
extern "C" void kernel_launch(void* const* d_in, const int* in_sizes, int n_in,
                              void* d_out, int out_size, void* d_ws, size_t ws_size,
                              hipStream_t stream)
{
    const int*   x     = (const int*)  d_in[0];
    const float* embw  = (const float*)d_in[2];
    const float* convw = (const float*)d_in[3];
    const float* convb = (const float*)d_in[4];
    const float* u4w   = (const float*)d_in[5];
    const float* gcnw  = (const float*)d_in[6];
    const float* gcnb  = (const float*)d_in[7];
    const float* adj   = (const float*)d_in[8];
    const float* f4tw  = (const float*)d_in[9];
    const float* f4tb  = (const float*)d_in[10];
    const float* f4w   = (const float*)d_in[11];
    const float* f4b   = (const float*)d_in[12];

    // workspace layout (all 16B aligned)
    char* ws = (char*)d_ws;
    const size_t off_hp  = 0;                                  // 2,560,000 B
    const size_t off_u4  = off_hp  + (size_t)Bb * Ls * 64 * 2;
    const size_t off_m4t = off_u4  + (size_t)Ys * 64 * 2;      // 1,142,016 B
    const size_t off_sb  = off_m4t + (size_t)Bb * Ys * Fs * 4; // 14,275,200 B
    const size_t off_o1  = off_sb  + (size_t)NCOL * KPAD * 2;  // 7,142,400 B
    bf16_t* hp  = (bf16_t*)(ws + off_hp);
    bf16_t* u4b = (bf16_t*)(ws + off_u4);
    float*  m4t = (float*) (ws + off_m4t);
    bf16_t* SbT = (bf16_t*)(ws + off_sb);
    float*  o1  = (float*) (ws + off_o1);

    embconv_kernel<<<dim3(40, Bb), 256, 0, stream>>>(x, embw, convw, convb, hp);
    u4cvt_kernel  <<<(Ys * 64 + 255) / 256, 256, 0, stream>>>(u4w, u4b);
    attn_kernel   <<<dim3((NYT + 3) / 4, Bb), 128, 0, stream>>>(hp, u4b, m4t);
    support_kernel<<<(NCOL * KPAD) / 256, 256, 0, stream>>>(m4t, gcnw, SbT);
    adjgemm_kernel<<<NYT, 128, 0, stream>>>(adj, SbT, o1);
    final_kernel  <<<(Bb * Ys + 255) / 256, 256, 0, stream>>>(
        m4t, o1, gcnb, f4tw, f4tb, f4w, f4b, (float*)d_out);
}
